// MultiHeadAttention_56942676411117
// MI455X (gfx1250) — matrix operand order repaired and checked by reading, not compiled, not run on hardware
//
#include <hip/hip_runtime.h>
#include <hip/hip_bf16.h>
#include <stdint.h>

// ---------------- problem constants ----------------
#define B_    2
#define T_    2048
#define DIN   1024
#define EMB_  1024
#define H_    16
#define HD_   64
#define MTOT  (B_ * T_)        // 4096 rows

typedef __attribute__((ext_vector_type(16))) __bf16 v16bf;
typedef __attribute__((ext_vector_type(8)))  float  v8f;
typedef __attribute__((ext_vector_type(4)))  unsigned int u32x4;
typedef __attribute__((ext_vector_type(8)))  int i32x8;
typedef __attribute__((ext_vector_type(4)))  int i32x4;

#if defined(__has_builtin)
#if __has_builtin(__builtin_amdgcn_tensor_load_to_lds) && __has_builtin(__builtin_amdgcn_s_wait_tensorcnt)
#define HAVE_TDM 1
#endif
#endif

// ---------------- helpers ----------------
__device__ __forceinline__ unsigned short f32_to_bf16(float f) {
  union { float f; uint32_t u; } c; c.f = f;
  uint32_t u = c.u;
  return (unsigned short)((u + 0x7FFFu + ((u >> 16) & 1u)) >> 16);
}

// Build a v16bf fragment from two aligned 16-byte LDS chunks.
__device__ __forceinline__ v16bf ldsv16(const unsigned short* p0, const unsigned short* p1) {
  union { v16bf v; uint4 q[2]; } u;
  u.q[0] = *(const uint4*)p0;
  u.q[1] = *(const uint4*)p1;
  return u.v;
}

// ---------------- Tensor Data Mover: 2D tile load Global -> LDS ----------------
// D# per CDNA5 ISA. One descriptor, bf16 (data_size=2B),
// LDS row padding (pad_interval: 64B->3, 128B->4; pad_amount 3 -> 4 DWORDs = 16B).
// Issued by the calling wave (EXEC ignored by TDM). Fallback: wave-level copy.
__device__ __forceinline__ void tdm_load_2d(const void* gptr, unsigned short* ldsPtr,
                                            uint32_t width, uint32_t height,
                                            uint32_t row_stride_elems,
                                            uint32_t pad_interval, uint32_t pad_amount) {
#ifdef HAVE_TDM
  uint64_t ga = (uint64_t)(uintptr_t)gptr;
  uint32_t lds_byte_off = (uint32_t)(uintptr_t)ldsPtr;  // LDS aperture: addr[31:0] = offset
  u32x4 g0 = { 1u /*count=1*/, lds_byte_off, (uint32_t)ga,
               (((uint32_t)(ga >> 32)) & 0x01FFFFFFu) | 0x80000000u /*type=2*/ };
  uint32_t w0 = (1u << 16) /*data_size=2B*/
              | (1u << 20) /*pad_enable*/
              | (pad_interval << 22) | (pad_amount << 25);
  i32x8 g1 = { (int)w0,
               (int)((width & 0xFFFFu) << 16),                    // tensor_dim0 lo16
               (int)((width >> 16) | ((height & 0xFFFFu) << 16)), // dim0 hi | dim1 lo
               (int)((height >> 16) | ((width & 0xFFFFu) << 16)), // dim1 hi | tile_dim0
               (int)(height & 0xFFFFu),                           // tile_dim1 | tile_dim2=0
               (int)row_stride_elems,                             // tensor_dim0_stride lo32
               0, 0 };
  i32x4 z4 = { 0, 0, 0, 0 };
#if __clang_major__ >= 23
  i32x8 z8 = { 0, 0, 0, 0, 0, 0, 0, 0 };
  __builtin_amdgcn_tensor_load_to_lds(g0, g1, z4, z4, z8, 0);
#else
  __builtin_amdgcn_tensor_load_to_lds(g0, g1, z4, z4, 0);
#endif
#else
  // Fallback: lane-parallel copy by the calling wave, same padded dst stride.
  const unsigned short* src = (const unsigned short*)gptr;
  const uint32_t dstStride = width + (pad_amount + 1u) * 2u;  // pad DWORDs -> elems
  const uint32_t lane = threadIdx.x & 31u;
  for (uint32_t r = lane; r < height; r += 32u)
    for (uint32_t c = 0; c < width; c += 8u)
      *(uint4*)&ldsPtr[r * dstStride + c] =
          *(const uint4*)&src[(size_t)r * row_stride_elems + c];
  (void)pad_interval;
#endif
}

__device__ __forceinline__ void tdm_wait0() {
#ifdef HAVE_TDM
  __builtin_amdgcn_s_wait_tensorcnt(0);
#endif
}

// ---------------- fp32 -> bf16 convert (4-wide) ----------------
__global__ void __launch_bounds__(256) cvt_bf16_kernel(const float* __restrict__ in,
                                                       unsigned short* __restrict__ out, int n4) {
  int i = blockIdx.x * 256 + threadIdx.x;
  if (i < n4) {
    float4 f = ((const float4*)in)[i];
    ushort4 o;
    o.x = f32_to_bf16(f.x); o.y = f32_to_bf16(f.y);
    o.z = f32_to_bf16(f.z); o.w = f32_to_bf16(f.w);
    ((ushort4*)out)[i] = o;
  }
}

// ---------------- WMMA GEMM: C[M,N] = A[M,K] * B[K,N] + bias ----------------
#define BM 128
#define BN 64
#define BK 32
#define ALD 40   // 32 + 8 pad (TDM pad: 64B rows + 16B) -> 80B stride
#define BTLD 40  // transposed B: 32 K-elems + 8 pad -> 80B stride

__global__ void __launch_bounds__(256) gemm_bf16_kernel(
    const unsigned short* __restrict__ A,
    const unsigned short* __restrict__ Bm,
    const float* __restrict__ bias,
    int M, int N, int K, int mode,
    float* __restrict__ Cout,
    unsigned short* __restrict__ Qo,
    unsigned short* __restrict__ Ko,
    unsigned short* __restrict__ Vo) {
  __shared__ __align__(16) unsigned short Asm[2][BM * ALD];   // 20 KB (double)
  __shared__ __align__(16) unsigned short BsmT[2][BN * BTLD]; // 10 KB (double)

  const int tid  = threadIdx.x;
  const int wave = tid >> 5;
  const int lane = tid & 31;
  const int ln   = lane & 15;
  const int lh   = lane >> 4;

  const int bn0 = blockIdx.x * BN;
  const int bm0 = blockIdx.y * BM;

  const int wm = (wave & 3) * 32;   // 4 waves along M
  const int wn = (wave >> 2) * 32;  // 2 waves along N

  v8f acc[2][2];
#pragma unroll
  for (int im = 0; im < 2; ++im)
#pragma unroll
    for (int in = 0; in < 2; ++in) acc[im][in] = (v8f)0.0f;

  const int NT = K / BK;

  // ---- prologue: stage tile 0 ----
  if (wave == 0)
    tdm_load_2d(A + (size_t)bm0 * K, &Asm[0][0], BK, BM, (uint32_t)K, 3, 3);
  { // transposed B fill, tile 0
    int row = tid >> 3;                 // 0..31 (K)
    int c0  = (tid & 7) * 8;            // 0..56 (N)
    union { uint4 v; unsigned short s[8]; } bu;
    bu.v = *(const uint4*)(Bm + (size_t)row * N + bn0 + c0);
#pragma unroll
    for (int j = 0; j < 8; ++j) BsmT[0][(c0 + j) * BTLD + row] = bu.s[j];
  }

  for (int it = 0; it < NT; ++it) {
    const int cur = it & 1;
    if (wave == 0) tdm_wait0();         // A tile `it` DMA complete
    __syncthreads();                    // tile `it` visible; compute(it-1) done

    if (it + 1 < NT) {
      const int nxt = cur ^ 1;
      const int k0n = (it + 1) * BK;
      if (wave == 0)
        tdm_load_2d(A + (size_t)bm0 * K + k0n, &Asm[nxt][0], BK, BM, (uint32_t)K, 3, 3);
      int row = tid >> 3;
      int c0  = (tid & 7) * 8;
      union { uint4 v; unsigned short s[8]; } bu;
      bu.v = *(const uint4*)(Bm + (size_t)(k0n + row) * N + bn0 + c0);
#pragma unroll
      for (int j = 0; j < 8; ++j) BsmT[nxt][(c0 + j) * BTLD + row] = bu.s[j];
    }

    // ---- compute tile `it` ----
    v16bf af[2];
#pragma unroll
    for (int im = 0; im < 2; ++im) {
      int row = wm + im * 16 + ln;
      const unsigned short* base = &Asm[cur][row * ALD + lh * 8];
      af[im] = ldsv16(base, base + 16);
    }
    v16bf bf2[2];
#pragma unroll
    for (int in = 0; in < 2; ++in) {
      const unsigned short* base = &BsmT[cur][(wn + in * 16 + ln) * BTLD + lh * 8];
      bf2[in] = ldsv16(base, base + 16);
    }
#pragma unroll
    for (int im = 0; im < 2; ++im)
#pragma unroll
      for (int in = 0; in < 2; ++in)
        acc[im][in] = __builtin_amdgcn_wmma_f32_16x16x32_bf16(
            false, af[im], false, bf2[in], (short)0, acc[im][in], false, false);
  }

  // epilogue (C layout: VGPR r -> row r + lh*8, lane -> col ln)
#pragma unroll
  for (int im = 0; im < 2; ++im)
#pragma unroll
    for (int in = 0; in < 2; ++in)
#pragma unroll
      for (int r = 0; r < 8; ++r) {
        int row = bm0 + wm + im * 16 + lh * 8 + r;
        int col = bn0 + wn + in * 16 + ln;
        float v = acc[im][in][r] + bias[col];
        if (mode == 0) {
          Cout[(size_t)row * N + col] = v;
        } else {
          int b      = row >> 11;          // / T_
          int t      = row & (T_ - 1);
          int head   = col / (3 * HD_);
          int within = col - head * (3 * HD_);
          int chunk  = within >> 6;        // 0=q 1=k 2=v
          int d      = within & (HD_ - 1);
          unsigned short* dst = (chunk == 0) ? Qo : (chunk == 1) ? Ko : Vo;
          dst[((size_t)(b * H_ + head) * T_ + t) * HD_ + d] = f32_to_bf16(v);
        }
      }
}

// ---------------- flash attention ----------------
#define AQ   128   // queries per block (8 waves x 16 rows)
#define AKV  64    // keys per LDS tile
#define QLD  72    // 64 + 8 pad (TDM pad: 128B rows + 16B) -> 144B stride
#define KLD  72
#define VTLD 72    // transposed V: 64 K-elems + 8 pad

__global__ void __launch_bounds__(256) attn_kernel(
    const unsigned short* __restrict__ Qb,
    const unsigned short* __restrict__ Kb,
    const unsigned short* __restrict__ Vb,
    unsigned short* __restrict__ Ob) {   // [B*T, EMB] bf16
  __shared__ __align__(16) unsigned short Qs[AQ * QLD];       // 18 KB
  __shared__ __align__(16) unsigned short Ks[AKV * KLD];      //  9 KB
  __shared__ __align__(16) unsigned short VsT[HD_ * VTLD];    //  9 KB
  __shared__ __align__(16) unsigned short Ps[8][16 * AKV];    // 16 KB

  const int bh = blockIdx.y;           // b*H + h
  const int b  = bh / H_;
  const int h  = bh - b * H_;
  const int q0 = blockIdx.x * AQ;

  const int tid  = threadIdx.x;
  const int wave = tid >> 5;
  const int lane = tid & 31;
  const int ln   = lane & 15;
  const int lh   = lane >> 4;

  const size_t headBase = (size_t)bh * T_ * HD_;

  // stage Q tile 128x64 via TDM (padded rows)
  if (wave == 0) {
    tdm_load_2d(Qb + headBase + (size_t)q0 * HD_, &Qs[0], HD_, AQ, HD_, 4, 3);
    tdm_wait0();
  }
  __syncthreads();

  // per-wave Q A-frags (2 k-steps of 32 over HD=64)
  const int qrow = wave * 16 + ln;
  v16bf qf[2];
#pragma unroll
  for (int ks = 0; ks < 2; ++ks) {
    const unsigned short* base = &Qs[qrow * QLD + ks * 32 + lh * 8];
    qf[ks] = ldsv16(base, base + 16);
  }

  float m_run[8], l_run[8];
#pragma unroll
  for (int r = 0; r < 8; ++r) { m_run[r] = -3.0e38f; l_run[r] = 0.0f; }
  v8f oacc[4];
#pragma unroll
  for (int dt = 0; dt < 4; ++dt) oacc[dt] = (v8f)0.0f;

  const float scale = 0.125f;  // 1/sqrt(64)

  for (int kv0 = 0; kv0 < T_; kv0 += AKV) {
    __syncthreads();                     // previous compute done
    if (wave == 0)
      tdm_load_2d(Kb + headBase + (size_t)kv0 * HD_, &Ks[0], HD_, AKV, HD_, 4, 3);
    // transposed V fill (all threads, overlaps K DMA)
    for (int c = tid; c < AKV * HD_ / 8; c += 256) {
      int row = c >> 3, c0 = (c & 7) * 8;
      union { uint4 v; unsigned short s[8]; } vu;
      vu.v = *(const uint4*)(&Vb[headBase + (size_t)(kv0 + row) * HD_ + c0]);
#pragma unroll
      for (int j = 0; j < 8; ++j) VsT[(c0 + j) * VTLD + row] = vu.s[j];
    }
    if (wave == 0) tdm_wait0();
    __syncthreads();                     // K + V tiles visible

    // scores: S = Q * K^T  (4 n-tiles of 16 keys, 2 k-steps)
    v8f sacc[4];
#pragma unroll
    for (int nt = 0; nt < 4; ++nt) sacc[nt] = (v8f)0.0f;
#pragma unroll
    for (int nt = 0; nt < 4; ++nt) {
      int key = nt * 16 + ln;
#pragma unroll
      for (int ks = 0; ks < 2; ++ks) {
        const unsigned short* base = &Ks[key * KLD + ks * 32 + lh * 8];
        v16bf kfr = ldsv16(base, base + 16);
        sacc[nt] = __builtin_amdgcn_wmma_f32_16x16x32_bf16(
            false, qf[ks], false, kfr, (short)0, sacc[nt], false, false);
      }
    }

    // online softmax per row (row = r + lh*8)
    float alpha[8];
#pragma unroll
    for (int r = 0; r < 8; ++r) {
      float mx = -3.0e38f;
#pragma unroll
      for (int nt = 0; nt < 4; ++nt) {
        sacc[nt][r] *= scale;
        mx = fmaxf(mx, sacc[nt][r]);
      }
#pragma unroll
      for (int off = 1; off < 16; off <<= 1)
        mx = fmaxf(mx, __shfl_xor(mx, off, 32));
      float mnew = fmaxf(m_run[r], mx);
      float a = __expf(m_run[r] - mnew);
      m_run[r] = mnew;
      float s = 0.0f;
#pragma unroll
      for (int nt = 0; nt < 4; ++nt) {
        float p = __expf(sacc[nt][r] - mnew);
        sacc[nt][r] = p;
        s += p;
      }
#pragma unroll
      for (int off = 1; off < 16; off <<= 1)
        s += __shfl_xor(s, off, 32);
      l_run[r] = l_run[r] * a + s;
      alpha[r] = a;
    }
#pragma unroll
    for (int dt = 0; dt < 4; ++dt)
#pragma unroll
      for (int r = 0; r < 8; ++r) oacc[dt][r] *= alpha[r];

    // re-shape P (C layout) -> A layout via per-wave LDS patch
#pragma unroll
    for (int nt = 0; nt < 4; ++nt)
#pragma unroll
      for (int r = 0; r < 8; ++r)
        Ps[wave][(r + lh * 8) * AKV + nt * 16 + ln] = f32_to_bf16(sacc[nt][r]);
    // same-wave LDS RAW: compiler inserts s_wait_dscnt

    // O += P * V   (2 k-steps over 64 keys, 4 d-tiles)
#pragma unroll
    for (int ks = 0; ks < 2; ++ks) {
      const unsigned short* pbase = &Ps[wave][ln * AKV + ks * 32 + lh * 8];
      v16bf pf = ldsv16(pbase, pbase + 16);
#pragma unroll
      for (int dt = 0; dt < 4; ++dt) {
        const unsigned short* vbase = &VsT[(dt * 16 + ln) * VTLD + ks * 32 + lh * 8];
        v16bf vfr = ldsv16(vbase, vbase + 16);
        oacc[dt] = __builtin_amdgcn_wmma_f32_16x16x32_bf16(
            false, pf, false, vfr, (short)0, oacc[dt], false, false);
      }
    }
  }

  // normalize + store bf16 into [B*T, EMB] at col h*64 + d
  const size_t orowbase = (size_t)(b * T_ + q0) * EMB_ + (size_t)h * HD_;
#pragma unroll
  for (int dt = 0; dt < 4; ++dt)
#pragma unroll
    for (int r = 0; r < 8; ++r) {
      int row = wave * 16 + lh * 8 + r;
      float v = oacc[dt][r] / l_run[r];
      Ob[orowbase + (size_t)row * EMB_ + dt * 16 + ln] = f32_to_bf16(v);
    }
}

// ---------------- launch ----------------
extern "C" void kernel_launch(void* const* d_in, const int* in_sizes, int n_in,
                              void* d_out, int out_size, void* d_ws, size_t ws_size,
                              hipStream_t stream) {
  const float* x     = (const float*)d_in[0];
  const float* W_qkv = (const float*)d_in[1];
  const float* b_qkv = (const float*)d_in[2];
  const float* W_o   = (const float*)d_in[3];
  const float* b_o   = (const float*)d_in[4];
  float* out = (float*)d_out;

  char* ws = (char*)d_ws;
  size_t off = 0;
  unsigned short* Xb    = (unsigned short*)(ws + off); off += (size_t)MTOT * DIN * 2;
  unsigned short* Wqkvb = (unsigned short*)(ws + off); off += (size_t)DIN * 3 * EMB_ * 2;
  unsigned short* Wob   = (unsigned short*)(ws + off); off += (size_t)EMB_ * EMB_ * 2;
  unsigned short* Qb    = (unsigned short*)(ws + off); off += (size_t)MTOT * EMB_ * 2;
  unsigned short* Kb    = (unsigned short*)(ws + off); off += (size_t)MTOT * EMB_ * 2;
  unsigned short* Vb    = (unsigned short*)(ws + off); off += (size_t)MTOT * EMB_ * 2;
  unsigned short* Ob    = (unsigned short*)(ws + off); off += (size_t)MTOT * EMB_ * 2;
  (void)ws_size; (void)in_sizes; (void)n_in; (void)out_size;

  // 1) fp32 -> bf16
  {
    int n4 = MTOT * DIN / 4;
    cvt_bf16_kernel<<<(n4 + 255) / 256, 256, 0, stream>>>(x, Xb, n4);
    n4 = DIN * 3 * EMB_ / 4;
    cvt_bf16_kernel<<<(n4 + 255) / 256, 256, 0, stream>>>(W_qkv, Wqkvb, n4);
    n4 = EMB_ * EMB_ / 4;
    cvt_bf16_kernel<<<(n4 + 255) / 256, 256, 0, stream>>>(W_o, Wob, n4);
  }

  // 2) QKV projection: [4096,1024] x [1024,3072] -> scatter q/k/v bf16
  {
    dim3 grid(3 * EMB_ / BN, MTOT / BM);
    gemm_bf16_kernel<<<grid, 256, 0, stream>>>(Xb, Wqkvb, b_qkv,
                                               MTOT, 3 * EMB_, DIN, /*mode=*/1,
                                               nullptr, Qb, Kb, Vb);
  }

  // 3) flash attention -> Ob bf16 [B*T, EMB]
  {
    dim3 grid(T_ / AQ, B_ * H_);
    attn_kernel<<<grid, 256, 0, stream>>>(Qb, Kb, Vb, Ob);
  }

  // 4) output projection: [4096,1024] x [1024,1024] + b_o -> fp32 d_out
  {
    dim3 grid(EMB_ / BN, MTOT / BM);
    gemm_bf16_kernel<<<grid, 256, 0, stream>>>(Ob, Wob, b_o,
                                               MTOT, EMB_, EMB_, /*mode=*/0,
                                               out, nullptr, nullptr, nullptr);
  }
}
